// BDH_GPU_2087354106305
// MI455X (gfx1250) — matrix operand order repaired and checked by reading, not compiled, not run on hardware
//
#include <hip/hip_runtime.h>
#include <hip/hip_bf16.h>

// Problem constants (match reference)
#define DN 4096      // N
#define DD 256       // D
#define DV 32000     // V
#define DB 32        // B
#define DT 256       // T
#define DM (DB*DT)   // 8192 rows total

typedef __attribute__((ext_vector_type(16))) _Float16 v16h;
typedef __attribute__((ext_vector_type(8)))  float    v8f;

// ------------------------------------------------ async copy (CDNA5 ASYNCcnt)
#if defined(__AMDGCN__) && __has_builtin(__builtin_amdgcn_global_load_async_to_lds_b128)
#define ASYNC_COPY 1
#else
#define ASYNC_COPY 0
#endif

typedef int g4i __attribute__((vector_size(16)));   // matches builtin param pointee

// Always declared (host pass parses discarded constexpr branches).
__device__ __forceinline__ void async_ld_b128(const _Float16* g, _Float16* l) {
#if ASYNC_COPY
    __builtin_amdgcn_global_load_async_to_lds_b128(
        (__attribute__((address_space(1))) g4i*)(__attribute__((address_space(1))) void*)(void*)g,
        (__attribute__((address_space(3))) g4i*)(__attribute__((address_space(3))) void*)(void*)l,
        0, 0);
#else
    (void)g; (void)l;
#endif
}

template <int N>
__device__ __forceinline__ void wait_asynccnt() {
#if defined(__AMDGCN__)
#if __has_builtin(__builtin_amdgcn_s_wait_asynccnt)
    __builtin_amdgcn_s_wait_asynccnt(N);
#else
    asm volatile("s_wait_asynccnt %0" ::"i"(N));
#endif
#endif
}

// ---------------------------------------------------------------- reductions
__device__ __forceinline__ float blockSum256(float v, float* s) {
    int tid = threadIdx.x;
#pragma unroll
    for (int o = 16; o > 0; o >>= 1) v += __shfl_xor(v, o, 32);
    __syncthreads();
    if ((tid & 31) == 0) s[tid >> 5] = v;
    __syncthreads();
    float t = s[0] + s[1] + s[2] + s[3] + s[4] + s[5] + s[6] + s[7];
    return t;
}

// ------------------------------------------------------- elementwise convert
__global__ __launch_bounds__(256) void f32_to_f16(const float* __restrict__ in,
                                                  _Float16* __restrict__ out, int n) {
    int i = blockIdx.x * 256 + threadIdx.x;
    if (i < n) out[i] = (_Float16)in[i];
}

// ---------------------------------------------- gather + layernorm of tokens
__global__ __launch_bounds__(256) void gather_ln(const int* __restrict__ idx,
                                                 const float* __restrict__ emb,
                                                 _Float16* __restrict__ Vh,
                                                 _Float16* __restrict__ Uh) {
    __shared__ float s[8];
    int m = blockIdx.x, d = threadIdx.x;
    int tok = idx[m];
    float v = emb[(size_t)tok * DD + d];
    Vh[(size_t)m * DD + d] = (_Float16)v;
    float mean = blockSum256(v, s) * (1.0f / DD);
    float c = v - mean;
    float var = blockSum256(c * c, s) * (1.0f / (DD - 1));
    float o = c / (sqrtf(var) + 1e-6f);
    Uh[(size_t)m * DD + d] = (_Float16)o;
}

// -------------------------------------------------------- row layernorm (D)
template <bool OUT16>
__global__ __launch_bounds__(256) void ln_kernel(const float* __restrict__ in,
                                                 void* __restrict__ out) {
    __shared__ float s[8];
    int m = blockIdx.x, d = threadIdx.x;
    float v = in[(size_t)m * DD + d];
    float mean = blockSum256(v, s) * (1.0f / DD);
    float c = v - mean;
    float var = blockSum256(c * c, s) * (1.0f / (DD - 1));
    float o = c / (sqrtf(var) + 1e-6f);
    if (OUT16) ((_Float16*)out)[(size_t)m * DD + d] = (_Float16)o;
    else       ((float*)out)[(size_t)m * DD + d] = o;
}

// ------------------------------------- sequential x-scan with L1 norm per t
__global__ __launch_bounds__(1024) void scan_kernel(float* __restrict__ X,
                                                    _Float16* __restrict__ Xh) {
    __shared__ float s[32];
    int b = blockIdx.x, tid = threadIdx.x;
    float xv[4] = {0.f, 0.f, 0.f, 0.f};
    for (int t = 0; t < DT; ++t) {
        float* row = X + (size_t)(b * DT + t) * DN;     // holds relu(v@Dx^T)
        float loc = 0.f;
#pragma unroll
        for (int k = 0; k < 4; ++k) {
            int n = tid + k * 1024;
            float x = 0.97f * xv[k] + row[n];
            xv[k] = x;
            loc += fabsf(x);
        }
#pragma unroll
        for (int o = 16; o > 0; o >>= 1) loc += __shfl_xor(loc, o, 32);
        __syncthreads();
        if ((tid & 31) == 0) s[tid >> 5] = loc;
        __syncthreads();
        float tot = 0.f;
#pragma unroll
        for (int w = 0; w < 32; ++w) tot += s[w];
        float inv = 1.0f / (tot + 1e-6f);
        _Float16* rowh = Xh + (size_t)(b * DT + t) * DN;
#pragma unroll
        for (int k = 0; k < 4; ++k) {
            int n = tid + k * 1024;
            float x = xv[k] * inv;
            xv[k] = x;
            row[n] = x;
            rowh[n] = (_Float16)x;
        }
    }
}

// -------------------------------------------------------------- WMMA GEMM
// C(MxN) = epilogue( A(MxK,f16,row) * B^T )   B: NT => (NxK,row), NN => (KxN,row)
// 128x128 block tile, BK=32 (== one WMMA K), 8 waves of 64x32 (4x2 subtiles).
// Double-buffered LDS with async global->LDS copies (ASYNCcnt pipeline).
#define BM 128
#define BN 128
#define BK 32
#define LDT 40   // padded LDS row (halves)

// EPI: 0 plain f32, 1 relu f32, 2 decay-mask f16 (scores), 3 relu*relu(X) f16
template <int EPI, bool BNN>
__global__ __launch_bounds__(256) void gemm_wmma(
    const _Float16* __restrict__ Ag, int lda, long long sA,
    const _Float16* __restrict__ Bg, int ldb, long long sB,
    void* __restrict__ Cv, int ldc, long long sC,
    int K, const float* __restrict__ Xm, long long sX) {
    __shared__ __align__(16) _Float16 ldsA[2][BM * LDT];
    __shared__ __align__(16) _Float16 ldsB[2][BN * LDT];

    const int tid = threadIdx.x;
    const int z = blockIdx.z;
    Ag += (size_t)z * sA;
    Bg += (size_t)z * sB;
    const int mBase = blockIdx.y * BM;
    const int nBase = blockIdx.x * BN;
    const int wid = tid >> 5, lane = tid & 31;
    const int wm = wid & 1, wn = wid >> 1;   // 2 x 4 wave grid
    const int halfSel = lane >> 4, l16 = lane & 15;

    v8f zero = {};
    v8f acc[4][2];
#pragma unroll
    for (int i = 0; i < 4; ++i)
#pragma unroll
        for (int j = 0; j < 2; ++j) acc[i][j] = zero;

    // one 16x16x32 MACC sweep over a staged tile pair
    auto compute = [&](const _Float16* tA, const _Float16* tB) {
        v16h af[4], bf[2];
#pragma unroll
        for (int i = 0; i < 4; ++i) {
            const _Float16* p = tA + (wm * 64 + i * 16 + l16) * LDT + halfSel * 16;
#pragma unroll
            for (int q = 0; q < 8; ++q) af[i][q] = p[q];
#pragma unroll
            for (int q = 0; q < 8; ++q) af[i][8 + q] = p[8 + q];
        }
#pragma unroll
        for (int j = 0; j < 2; ++j) {
            const _Float16* p = tB + (wn * 32 + j * 16 + l16) * LDT + halfSel * 16;
#pragma unroll
            for (int q = 0; q < 16; ++q) bf[j][q] = p[q];
        }
#pragma unroll
        for (int i = 0; i < 4; ++i)
#pragma unroll
            for (int j = 0; j < 2; ++j)
                acc[i][j] = __builtin_amdgcn_wmma_f32_16x16x32_f16(
                    false, af[i], false, bf[j], (short)0, acc[i][j], false, false);
    };

    if constexpr (!BNN && ASYNC_COPY) {
        // ---- async double-buffered pipeline: 1024 16B chunks/tile-pair,
        //      2 chunks per array per thread => 4 async issues/wave/step.
        int crow[2], cseg[2];
#pragma unroll
        for (int i = 0; i < 2; ++i) {
            int c = tid + i * 256;          // 0..511
            crow[i] = c >> 2;               // 0..127
            cseg[i] = (c & 3) * 8;          // halves: 0,8,16,24
        }
        auto issue = [&](int kk, int buf) {
#pragma unroll
            for (int i = 0; i < 2; ++i)
                async_ld_b128(Ag + (size_t)(mBase + crow[i]) * lda + kk + cseg[i],
                              &ldsA[buf][crow[i] * LDT + cseg[i]]);
#pragma unroll
            for (int i = 0; i < 2; ++i)
                async_ld_b128(Bg + (size_t)(nBase + crow[i]) * ldb + kk + cseg[i],
                              &ldsB[buf][crow[i] * LDT + cseg[i]]);
        };
        const int steps = K / BK;
        issue(0, 0);
        for (int k = 0; k < steps; ++k) {
            const int cur = k & 1;
            if (k + 1 < steps) {
                issue((k + 1) * BK, cur ^ 1);   // prefetch next tile
                wait_asynccnt<4>();             // older 4 (= tile k) complete in order
            } else {
                wait_asynccnt<0>();
            }
            __syncthreads();                    // all waves' portions landed
            compute(ldsA[cur], ldsB[cur]);
            __syncthreads();                    // done reading before next overwrite
        }
    } else {
        // ---- synchronous fallback (and NN-transpose staging for S@U)
        const int row = tid >> 1;            // 0..127
        const int kseg = (tid & 1) * 16;     // 0 or 16
        for (int k0 = 0; k0 < K; k0 += BK) {
            {
                const _Float16* src = Ag + (size_t)(mBase + row) * lda + k0 + kseg;
                _Float16* dst = &ldsA[0][row * LDT + kseg];
                *(float4*)dst       = *(const float4*)src;
                *(float4*)(dst + 8) = *(const float4*)(src + 8);
            }
            if constexpr (!BNN) {
                const _Float16* src = Bg + (size_t)(nBase + row) * ldb + k0 + kseg;
                _Float16* dst = &ldsB[0][row * LDT + kseg];
                *(float4*)dst       = *(const float4*)src;
                *(float4*)(dst + 8) = *(const float4*)(src + 8);
            } else {    // B stored KxN: transpose into [n][k]
#pragma unroll
                for (int i = 0; i < 16; ++i) {
                    int e = tid * 16 + i;
                    int n = e >> 5, k = e & 31;
                    ldsB[0][n * LDT + k] = Bg[(size_t)(k0 + k) * ldb + nBase + n];
                }
            }
            __syncthreads();
            compute(ldsA[0], ldsB[0]);
            __syncthreads();
        }
    }

    // epilogue: C 16x16 f32 layout — VGPR v holds row v + 8*(lane>=16), col = lane&15
#pragma unroll
    for (int i = 0; i < 4; ++i) {
#pragma unroll
        for (int j = 0; j < 2; ++j) {
            int gr0 = mBase + wm * 64 + i * 16 + halfSel * 8;
            int gc = nBase + wn * 32 + j * 16 + l16;
#pragma unroll
            for (int v = 0; v < 8; ++v) {
                int gr = gr0 + v;
                float val = acc[i][j][v];
                size_t off = (size_t)gr * ldc + gc;
                if constexpr (EPI == 0) {
                    ((float*)Cv)[(size_t)z * sC + off] = val;
                } else if constexpr (EPI == 1) {
                    ((float*)Cv)[(size_t)z * sC + off] = fmaxf(val, 0.0f);
                } else if constexpr (EPI == 2) {
                    // decay mask: weight = 0.97^(t-s) for s<t else 0 (rho_{t-1} readout)
                    float w = (gc < gr) ? __expf((float)(gr - gc) * -0.030459207f) : 0.0f;
                    ((_Float16*)Cv)[(size_t)z * sC + off] = (_Float16)(val * w);
                } else {
                    float xv = Xm[(size_t)z * sX + off];
                    ((_Float16*)Cv)[(size_t)z * sC + off] =
                        (_Float16)(fmaxf(val, 0.0f) * fmaxf(xv, 0.0f));
                }
            }
        }
    }
}

// ------------------------------------------------------------------- driver
extern "C" void kernel_launch(void* const* d_in, const int* in_sizes, int n_in,
                              void* d_out, int out_size, void* d_ws, size_t ws_size,
                              hipStream_t stream) {
    const int*   idx = (const int*)d_in[0];
    const float* emb = (const float*)d_in[1];   // (V,D)
    const float* E   = (const float*)d_in[2];   // (D,N)
    const float* Dx  = (const float*)d_in[3];   // (N,D)
    const float* Dy  = (const float*)d_in[4];   // (N,D)
    float* out = (float*)d_out;                 // (B,T,D)

    char* ws = (char*)d_ws;
    size_t off = 0;
    auto alloc = [&](size_t bytes) { size_t o = off; off = (off + bytes + 255) & ~(size_t)255; return o; };

    _Float16* Vh   = (_Float16*)(ws + alloc((size_t)DM * DD * 2));
    _Float16* Uh   = (_Float16*)(ws + alloc((size_t)DM * DD * 2));
    _Float16* Dxh  = (_Float16*)(ws + alloc((size_t)DN * DD * 2));
    _Float16* Dyh  = (_Float16*)(ws + alloc((size_t)DN * DD * 2));
    _Float16* Eh   = (_Float16*)(ws + alloc((size_t)DD * DN * 2));
    float*    X    = (float*)   (ws + alloc((size_t)DM * DN * 4));  // R then x_t
    _Float16* Xh   = (_Float16*)(ws + alloc((size_t)DM * DN * 2));  // reused as Yh
    _Float16* Sh   = (_Float16*)(ws + alloc((size_t)DB * DT * DT * 2));
    float*    As   = (float*)   (ws + alloc((size_t)DM * DD * 4));  // A*, reused as Z
    _Float16* Ah   = (_Float16*)(ws + alloc((size_t)DM * DD * 2));
    _Float16* Yh   = Xh;   // Xh dead after scores
    float*    Z    = As;   // A* dead after its layernorm

    const int nw = DN * DD;
    f32_to_f16<<<(nw + 255) / 256, 256, 0, stream>>>(Dx, Dxh, nw);
    f32_to_f16<<<(nw + 255) / 256, 256, 0, stream>>>(Dy, Dyh, nw);
    f32_to_f16<<<(nw + 255) / 256, 256, 0, stream>>>(E, Eh, nw);

    // 1. gather tokens + layernorm
    gather_ln<<<DM, 256, 0, stream>>>(idx, emb, Vh, Uh);

    // 2. R = relu(V @ Dx^T)     (8192 x 4096, K=256)
    gemm_wmma<1, false><<<dim3(DN / BN, DM / BM, 1), 256, 0, stream>>>(
        Vh, DD, 0, Dxh, DD, 0, (void*)X, DN, 0, DD, nullptr, 0);

    // 3. sequential scan + L1 normalize (writes X f32 in place + Xh f16)
    scan_kernel<<<DB, 1024, 0, stream>>>(X, Xh);

    // 4. per-batch scores S = decay_mask(X X^T)   (256 x 256, K=4096) x 32
    gemm_wmma<2, false><<<dim3(DT / BN, DT / BM, DB), 256, 0, stream>>>(
        Xh, DN, (long long)DT * DN, Xh, DN, (long long)DT * DN,
        (void*)Sh, DT, (long long)DT * DT, DN, nullptr, 0);

    // 5. A* = S @ U             (256 x 256, K=256, NN) x 32
    gemm_wmma<0, true><<<dim3(DD / BN, DT / BM, DB), 256, 0, stream>>>(
        Sh, DT, (long long)DT * DT, Uh, DD, (long long)DT * DD,
        (void*)As, DD, (long long)DT * DD, DT, nullptr, 0);

    // 6. Ah = layernorm(A*) in f16
    ln_kernel<true><<<DM, 256, 0, stream>>>(As, (void*)Ah);

    // 7. Y = relu(Ah @ Dy^T) * relu(X)   (8192 x 4096, K=256)
    gemm_wmma<3, false><<<dim3(DN / BN, DM / BM, 1), 256, 0, stream>>>(
        Ah, DD, 0, Dyh, DD, 0, (void*)Yh, DN, 0, DD, X, 0);

    // 8. Z = Y @ E^T            (8192 x 256, K=4096)
    gemm_wmma<0, false><<<dim3(DD / BN, DM / BM, 1), 256, 0, stream>>>(
        Yh, DN, 0, Eh, DN, 0, (void*)Z, DD, 0, DN, nullptr, 0);

    // 9. out = layernorm(Z)  f32
    ln_kernel<false><<<DM, 256, 0, stream>>>(Z, (void*)out);
}
